// RSNEncoder_28845000360760
// MI455X (gfx1250) — matrix operand order repaired and checked by reading, not compiled, not run on hardware
//
#include <hip/hip_runtime.h>

// ---------------------------------------------------------------------------
// CDNA5 (gfx1250) bf16-WMMA GRU encoder, TDM-staged A tiles.
//   S=8, B=32768, D=512. All GEMMs K=512, bf16 A/B, f32 accumulate.
// ---------------------------------------------------------------------------

typedef __attribute__((ext_vector_type(16))) __bf16 v16bf;
typedef __attribute__((ext_vector_type(8)))  float  v8f;

union Frag16 {
    v16bf v;
    uint4 q[2];
};

__device__ __forceinline__ unsigned short f32_to_bf16(float f) {
    unsigned int u = __float_as_uint(f);
    unsigned int r = u + 0x7FFFu + ((u >> 16) & 1u);   // round-to-nearest-even
    return (unsigned short)(r >> 16);
}

// ---------------------------------------------------------------------------
// TDM descriptor for one 128(row) x 32(K) bf16 tile, row stride 512 elems,
// LDS rows padded to 80B (pad_interval: 16 DWORDs -> code 3; pad: 4 DWORDs -> 3)
// ---------------------------------------------------------------------------
#if __has_builtin(__builtin_amdgcn_tensor_load_to_lds)
#define RSN_USE_TDM 1
typedef unsigned int u32x4 __attribute__((ext_vector_type(4)));
typedef int          i32x8 __attribute__((ext_vector_type(8)));
typedef int          i32x4 __attribute__((ext_vector_type(4)));

__device__ __forceinline__ void tdm_load_tile_128x32(const unsigned short* gsrc,
                                                     unsigned int ldsOff, int M)
{
    const unsigned long long ga = (unsigned long long)(uintptr_t)gsrc;
    u32x4 g0 = {
        1u,                                          // count=1 valid user D#
        ldsOff,                                      // lds_addr (bytes)
        (unsigned)(ga & 0xFFFFFFFFu),                // global_addr[31:0]
        (unsigned)((ga >> 32) & 0x01FFFFFFu) | (2u << 30)  // addr[56:32] | type=2
    };
    i32x8 g1 = {
        (int)((1u << 16) | (1u << 20) | (3u << 22) | (3u << 25)), // 2B elems, pad 4dw/16dw
        (int)(512u << 16),                           // tensor_dim0 = 512 (bits 63:48)
        (int)(((unsigned)M & 0xFFFFu) << 16),        // tensor_dim1[15:0]
        (int)((((unsigned)M >> 16) & 0xFFFFu) | (32u << 16)), // dim1 hi | tile_dim0=32
        128,                                         // tile_dim1 = 128 rows
        512,                                         // tensor_dim0_stride = 512
        0, 0
    };
    i32x4 gz4 = { 0, 0, 0, 0 };                      // 2D tensor: groups 2/3 unused
    i32x8 gz8 = { 0, 0, 0, 0, 0, 0, 0, 0 };
    __builtin_amdgcn_tensor_load_to_lds(g0, g1, gz4, gz4, gz8, 0);
}
#endif

// ---------------------------------------------------------------------------
// Per-K-step compute: 2 B-fragments from global bf16 weights, 4 A-fragments
// from LDS, 8 WMMAs. Layouts used (contiguous K, same for A and B):
//   A 16x32: lane grp0 K {0..15}, grp1 {16..31}  (2x ds_load_b128)
//   B 32x16: lanes 0-15 K=k0..k0+15 of col n, lanes 16-31 K=k0+16..31 (2x b128)
// ---------------------------------------------------------------------------
__device__ __forceinline__ void mma_tile(const unsigned short* __restrict__ buf,
                                         const unsigned short* __restrict__ W,
                                         int k0, int nBase, int wr, int wc,
                                         int l15, int lhi, v8f acc[4][2])
{
    constexpr int K = 512;
    constexpr int LSTR = 40;
    Frag16 bf[2];
#pragma unroll
    for (int nt = 0; nt < 2; ++nt) {
        const int n = nBase + wc * 32 + nt * 16 + l15;
        const uint4* wp = (const uint4*)(W + (size_t)n * K + k0 + lhi * 16);
        bf[nt].q[0] = wp[0];
        bf[nt].q[1] = wp[1];
    }
#pragma unroll
    for (int mt = 0; mt < 4; ++mt) {
        const int m = wr * 64 + mt * 16 + l15;
        const unsigned short* src = &buf[m * LSTR + lhi * 16];
        Frag16 af;
        af.q[0] = *(const uint4*)(src);        // K = kb .. kb+7
        af.q[1] = *(const uint4*)(src + 8);    // K = kb+8 .. kb+15
#pragma unroll
        for (int nt = 0; nt < 2; ++nt) {
            acc[mt][nt] = __builtin_amdgcn_wmma_f32_16x16x32_bf16(
                false, af.v, false, bf[nt].v, (short)0, acc[mt][nt], false, false);
        }
    }
}

// ---------------------------------------------------------------------------
// C[M,ldc] (=|+=) A[M,512](bf16) @ W[N,512](bf16)^T + bias; optional bf16 shadow.
// grid: (N/128, M/128), block: 256 threads (8 waves, 2x4 wave grid)
// ---------------------------------------------------------------------------
__global__ __launch_bounds__(256)
void gemm_bf16_wmma(const unsigned short* __restrict__ A,
                    const unsigned short* __restrict__ W,
                    const float* __restrict__ bias,
                    float* __restrict__ C,
                    unsigned short* __restrict__ Cb,
                    int ldc, int M, int accumulate)
{
    constexpr int K = 512;
    constexpr int LSTR = 40;                       // bf16/row: 32 data + 8 pad (80B)
    constexpr int TILE = 128 * LSTR;               // ushorts per LDS tile buffer
    constexpr int NK = K / 32;                     // 16 K-steps
    __shared__ __align__(16) unsigned short smA[2 * TILE];

    const int tid  = threadIdx.x;
    const int lane = tid & 31;
    const int wave = tid >> 5;
    const int wr   = wave >> 2;                    // 0..1 (M dir, 64 rows)
    const int wc   = wave & 3;                     // 0..3 (N dir, 32 cols)
    const int l15  = lane & 15;
    const int lhi  = lane >> 4;
    const int mBase = blockIdx.y * 128;
    const int nBase = blockIdx.x * 128;

    v8f acc[4][2] = {};

#if RSN_USE_TDM
    // ---- depth-2 TDM pipeline: DMA engine stages A tiles into padded LDS ----
    const unsigned int ldsBase = (unsigned int)(uintptr_t)&smA[0];
    const unsigned short* aTile = A + (size_t)mBase * K;
    if (wave == 0) {
        tdm_load_tile_128x32(aTile +  0, ldsBase,            M);
        tdm_load_tile_128x32(aTile + 32, ldsBase + TILE * 2, M);
    }
    for (int j = 0; j < NK; ++j) {
        const int k0 = j * 32;
        if (wave == 0) {
            if (j < NK - 1)
                __builtin_amdgcn_s_wait_tensorcnt(1);   // tile j retired, j+1 in flight
            else
                __builtin_amdgcn_s_wait_tensorcnt(0);   // final tile retired
        }
        __syncthreads();                            // tile j visible to all waves
        mma_tile(&smA[(j & 1) * TILE], W, k0, nBase, wr, wc, l15, lhi, acc);
        __syncthreads();                            // buf[j&1] free for reuse
        if (wave == 0 && j + 2 < NK)
            tdm_load_tile_128x32(aTile + k0 + 64, ldsBase + (j & 1) * TILE * 2, M);
    }
#else
    // ---- fallback: double-buffered cooperative VGPR staging ----
    const int srow  = tid >> 1;                     // 0..127
    const int shalf = (tid & 1) * 16;               // 0 or 16 (bf16 elems)
    const unsigned short* agp = A + (size_t)(mBase + srow) * K + shalf;
    uint4 st0 = ((const uint4*)agp)[0];
    uint4 st1 = ((const uint4*)agp)[1];
    for (int j = 0; j < NK; ++j) {
        const int k0 = j * 32;
        __syncthreads();                            // buf[j&1] free for overwrite
        unsigned short* buf = &smA[(j & 1) * TILE];
        uint4* dst = (uint4*)&buf[srow * LSTR + shalf];
        dst[0] = st0;
        dst[1] = st1;
        if (j + 1 < NK) {
            const unsigned short* nxt = agp + k0 + 32;
            st0 = ((const uint4*)nxt)[0];
            st1 = ((const uint4*)nxt)[1];
        }
        __syncthreads();                            // staging visible
        mma_tile(buf, W, k0, nBase, wr, wc, l15, lhi, acc);
    }
#endif

    // ---- epilogue: bias, optional accumulate, f32 (+ optional bf16) store ----
#pragma unroll
    for (int nt = 0; nt < 2; ++nt) {
        const int n = nBase + wc * 32 + nt * 16 + l15;
        const float bv = bias ? bias[n] : 0.0f;
#pragma unroll
        for (int mt = 0; mt < 4; ++mt) {
#pragma unroll
            for (int r = 0; r < 8; ++r) {
                const int m = mBase + wr * 64 + mt * 16 + r + lhi * 8;
                const size_t off = (size_t)m * ldc + n;
                float v = acc[mt][nt][r] + bv;
                if (accumulate) v += C[off];
                C[off] = v;
                if (Cb) Cb[off] = f32_to_bf16(v);
            }
        }
    }
}

// ---------------------------------------------------------------------------
// GRU gates: h = (1-z)*n + z*h  (gate order r,z,n); also emits bf16 shadow.
// ---------------------------------------------------------------------------
__global__ __launch_bounds__(256)
void gru_gate_kernel(const float* __restrict__ gi, const float* __restrict__ gh,
                     float* __restrict__ h, unsigned short* __restrict__ hb)
{
    const size_t idx = (size_t)blockIdx.x * blockDim.x + threadIdx.x;
    const size_t b = idx >> 9;
    const int    d = (int)(idx & 511);
    const size_t g = b * 1536 + d;

    const float ir = gi[g],        hr = gh[g];
    const float iz = gi[g +  512], hz = gh[g +  512];
    const float in = gi[g + 1024], hn = gh[g + 1024];

    const float r = 1.0f / (1.0f + __expf(-(ir + hr)));
    const float z = 1.0f / (1.0f + __expf(-(iz + hz)));
    const float n = tanhf(in + r * hn);

    const size_t ho = b * 512 + d;
    const float  hv = (1.0f - z) * n + z * h[ho];
    h[ho]  = hv;
    hb[ho] = f32_to_bf16(hv);
}

__global__ __launch_bounds__(256)
void f32_to_bf16_kernel(const float* __restrict__ src, unsigned short* __restrict__ dst)
{
    const size_t i = (size_t)blockIdx.x * blockDim.x + threadIdx.x;
    dst[i] = f32_to_bf16(src[i]);
}

__global__ __launch_bounds__(256)
void zero_f32_kernel(float* __restrict__ p)
{
    p[(size_t)blockIdx.x * blockDim.x + threadIdx.x] = 0.0f;
}

__global__ __launch_bounds__(256)
void zero_u16_kernel(unsigned short* __restrict__ p)
{
    p[(size_t)blockIdx.x * blockDim.x + threadIdx.x] = 0;
}

// ---------------------------------------------------------------------------

extern "C" void kernel_launch(void* const* d_in, const int* in_sizes, int n_in,
                              void* d_out, int out_size, void* d_ws, size_t ws_size,
                              hipStream_t stream)
{
    (void)in_sizes; (void)n_in; (void)out_size;

    const float* x   = (const float*)d_in[0];   // [8, 32768, 512]
    const float* Wih = (const float*)d_in[1];   // [1536, 512]
    const float* Whh = (const float*)d_in[2];   // [1536, 512]
    const float* bih = (const float*)d_in[3];   // [1536]
    const float* bhh = (const float*)d_in[4];   // [1536]
    const float* w1  = (const float*)d_in[5];   // [512, 512]
    const float* w2  = (const float*)d_in[6];   // [512, 512]
    float* out       = (float*)d_out;           // [32768, 512]

    const int S = 8, B = 32768, D = 512, N3 = 1536;

    // per-chunk ws bytes: 6144+6144+2048*2+1024*2+1024*2 = 20480 per row
    const size_t wbytes = (size_t)(2 * N3 * D + 2 * D * D) * 2 + 8192;
    int chunk = 128;
    for (int c = 32768; c >= 128; c >>= 1) {
        if (wbytes + (size_t)c * 20480 <= ws_size) { chunk = c; break; }
    }

    char* ws = (char*)d_ws;
    size_t off = 0;
    auto alloc = [&](size_t bytes) -> void* {
        void* p = ws + off;
        off += (bytes + 255) & ~(size_t)255;
        return p;
    };
    unsigned short* WihB = (unsigned short*)alloc((size_t)N3 * D * 2);
    unsigned short* WhhB = (unsigned short*)alloc((size_t)N3 * D * 2);
    unsigned short* w1B  = (unsigned short*)alloc((size_t)D * D * 2);
    unsigned short* w2B  = (unsigned short*)alloc((size_t)D * D * 2);
    float*          gi   = (float*)alloc((size_t)chunk * N3 * 4);
    float*          gh   = (float*)alloc((size_t)chunk * N3 * 4);
    float*          hF[2]  = { (float*)alloc((size_t)chunk * D * 4),
                               (float*)alloc((size_t)chunk * D * 4) };
    unsigned short* hX[2]  = { (unsigned short*)alloc((size_t)chunk * D * 2),
                               (unsigned short*)alloc((size_t)chunk * D * 2) };
    unsigned short* xb[2]  = { (unsigned short*)alloc((size_t)chunk * D * 2),
                               (unsigned short*)alloc((size_t)chunk * D * 2) };

    // one-time weight conversion to bf16
    f32_to_bf16_kernel<<<(N3 * D) / 256, 256, 0, stream>>>(Wih, WihB);
    f32_to_bf16_kernel<<<(N3 * D) / 256, 256, 0, stream>>>(Whh, WhhB);
    f32_to_bf16_kernel<<<(D * D) / 256, 256, 0, stream>>>(w1, w1B);
    f32_to_bf16_kernel<<<(D * D) / 256, 256, 0, stream>>>(w2, w2B);

    const dim3 blk(256);
    const dim3 grdG(N3 / 128, chunk / 128);
    const dim3 grdM(D / 128, chunk / 128);
    const int  eltBlocks = (chunk * D) / 256;

    for (int cb = 0; cb < B; cb += chunk) {
        float*          h  = hF[0];
        float*          tf = hF[1];
        unsigned short* hb = hX[0];
        unsigned short* tb = hX[1];
        zero_f32_kernel<<<eltBlocks, blk, 0, stream>>>(h);
        zero_u16_kernel<<<eltBlocks, blk, 0, stream>>>(hb);

        for (int i = 0; i < S; ++i) {
            // x_i -> bf16 (also serves as x_prev on the next step)
            f32_to_bf16_kernel<<<eltBlocks, blk, 0, stream>>>(
                x + ((size_t)i * B + cb) * D, xb[i & 1]);

            gemm_bf16_wmma<<<grdG, blk, 0, stream>>>(xb[i & 1], WihB, bih, gi,
                                                     nullptr, N3, chunk, 0);
            gemm_bf16_wmma<<<grdG, blk, 0, stream>>>(hb, WhhB, bhh, gh,
                                                     nullptr, N3, chunk, 0);
            gru_gate_kernel<<<eltBlocks, blk, 0, stream>>>(gi, gh, h, hb);

            if (i & 1) {  // mixed = h @ w1^T + x[i-1] @ w2^T
                const bool last = (i == S - 1);
                float*          mf = last ? (out + (size_t)cb * D) : tf;
                unsigned short* mb = last ? nullptr : tb;
                gemm_bf16_wmma<<<grdM, blk, 0, stream>>>(hb, w1B, nullptr, mf,
                                                         nullptr, D, chunk, 0);
                gemm_bf16_wmma<<<grdM, blk, 0, stream>>>(xb[(i - 1) & 1], w2B, nullptr,
                                                         mf, mb, D, chunk, 1);
                if (!last) {
                    float* t0 = h; h = mf; tf = t0;
                    unsigned short* t1 = hb; hb = mb; tb = t1;
                }
            }
        }
    }
}
